// LocalRNN_61366492725331
// MI455X (gfx1250) — matrix operand order repaired and checked by reading, not compiled, not run on hardware
//
#include <hip/hip_runtime.h>
#include <hip/hip_bf16.h>
#include <stdint.h>

// LocalRNN on MI455X (gfx1250, wave32, WMMA).
// B=8, L=2048, D=256, KSIZE=8.
// Phase 1: U = X @ W_in   (bf16 WMMA, f32 accumulate) -> d_ws (16384x256 f32)
// Phase 2: per 16-sequence tile, 8 recurrent steps of h@W on WMMA.
//          h lives in f32 registers in the C-matrix layout; a bf16 copy in LDS
//          feeds all waves' A-fragments. The 23 U rows a block ever needs are
//          staged into LDS once (zero-padded), killing 8x redundant U traffic.

#define BDIM   8
#define LDIM   2048
#define DDIM   256
#define KSTEPS 8
#define NROW   (BDIM * LDIM)   // 16384 independent sequences
#define UROWS  (16 + KSTEPS - 1)  // 23 staged U rows per block

typedef __attribute__((ext_vector_type(16))) __bf16 v16bf;
typedef __attribute__((ext_vector_type(8)))  float  v8f;

// Trivial 16-byte POD (HIP's uint4 has ctors -> can't live in a union).
struct alignas(16) B128 { unsigned int w[4]; };

union Frag {
    B128   q[2];   // two 16B halves = 8 VGPRs
    v16bf  v;      // WMMA operand view
    __bf16 e[16];  // element view for packing
};

// ---------------------------------------------------------------------------
// Kernel 1: U[16384 x 256] = X[16384 x 256] @ W_in[256 x 256]
// grid = (1024, 2), block = 256 (8 waves). Wave w handles N-tile blockIdx.y*8+w.
// ---------------------------------------------------------------------------
__global__ __launch_bounds__(256) void localrnn_proj_u(
    const float* __restrict__ x,
    const float* __restrict__ w_in,
    float* __restrict__ U)
{
    const int tid  = threadIdx.x;
    const int wv   = tid >> 5;
    const int lane = tid & 31;
    const int m    = lane & 15;   // A row within tile / C column within tile
    const int kh   = lane >> 4;   // lane half selects K sub-range
    const int ncol = (blockIdx.y * 8 + wv) * 16 + m;
    const long row0 = (long)blockIdx.x * 16;
    const long arow = row0 + m;

    v8f acc = {};
#pragma unroll
    for (int kc = 0; kc < 8; ++kc) {
        const int kbase = kc * 32 + kh * 8;
        // A fragment: x[arow][kbase..+7] -> e[0..7], x[arow][kbase+16..+23] -> e[8..15]
        Frag a;
        const float4 f0 = *reinterpret_cast<const float4*>(&x[arow * DDIM + kbase]);
        const float4 f1 = *reinterpret_cast<const float4*>(&x[arow * DDIM + kbase + 4]);
        const float4 f2 = *reinterpret_cast<const float4*>(&x[arow * DDIM + kbase + 16]);
        const float4 f3 = *reinterpret_cast<const float4*>(&x[arow * DDIM + kbase + 20]);
        a.e[0]  = (__bf16)f0.x; a.e[1]  = (__bf16)f0.y; a.e[2]  = (__bf16)f0.z; a.e[3]  = (__bf16)f0.w;
        a.e[4]  = (__bf16)f1.x; a.e[5]  = (__bf16)f1.y; a.e[6]  = (__bf16)f1.z; a.e[7]  = (__bf16)f1.w;
        a.e[8]  = (__bf16)f2.x; a.e[9]  = (__bf16)f2.y; a.e[10] = (__bf16)f2.z; a.e[11] = (__bf16)f2.w;
        a.e[12] = (__bf16)f3.x; a.e[13] = (__bf16)f3.y; a.e[14] = (__bf16)f3.z; a.e[15] = (__bf16)f3.w;

        // B fragment: W_in[k][ncol], lanes 0-15 cover K=base..+7,+16..+23, lanes 16-31 K=base+8..+15,+24..+31
        Frag b;
#pragma unroll
        for (int v = 0; v < 8; ++v) {
            const int k0 = kc * 32 + kh * 8 + 2 * v + (v >= 4 ? 8 : 0);
            b.e[2 * v]     = (__bf16)w_in[(long)k0 * DDIM + ncol];
            b.e[2 * v + 1] = (__bf16)w_in[(long)(k0 + 1) * DDIM + ncol];
        }
        acc = __builtin_amdgcn_wmma_f32_16x16x32_bf16(
            false, a.v, false, b.v, (short)0, acc, false, false);
    }

#pragma unroll
    for (int r = 0; r < 8; ++r) {
        const long orow = row0 + r + 8 * kh;  // C layout: lanes 0-15 M=0..7, lanes 16-31 M=8..15
        U[orow * DDIM + ncol] = acc[r];
    }
}

// ---------------------------------------------------------------------------
// Kernel 2: recurrence. grid = 1024 blocks (16 sequences each), block = 512
// (16 waves; wave w owns output columns [16w, 16w+16)).
// ---------------------------------------------------------------------------
__global__ __launch_bounds__(512) void localrnn_recur(
    const float* __restrict__ weight,
    const float* __restrict__ bias,
    const float* __restrict__ tau,
    const float* __restrict__ U,
    float* __restrict__ out)
{
    __shared__ __align__(16) __bf16 hsh[16 * DDIM];     // 8 KB: bf16 h for A-fragments
    __shared__ __align__(16) float  ush[UROWS * DDIM];  // 23.5 KB: staged U rows

    const int tid  = threadIdx.x;
    const int wv   = tid >> 5;
    const int lane = tid & 31;
    const int m    = lane & 15;
    const int kh   = lane >> 4;
    const int ncol = wv * 16 + m;
    const long row0 = (long)blockIdx.x * 16;
    const int  l0   = (int)(row0 & (LDIM - 1));  // tiles never cross a batch boundary

    // Stage U rows row0-7 .. row0+15 into LDS (zero-fill the pre-sequence pad).
    // 23 rows x 64 float4 = 1472 vector loads across 512 threads.
    for (int idx = tid; idx < UROWS * (DDIM / 4); idx += 512) {
        const int r  = idx >> 6;            // staged row 0..22 -> global row row0-7+r
        const int c4 = (idx & 63) << 2;     // float4 column
        float4 val = make_float4(0.f, 0.f, 0.f, 0.f);
        if (r >= KSTEPS - 1 - l0)           // only first tile of each batch pads
            val = *reinterpret_cast<const float4*>(
                &U[(row0 - (KSTEPS - 1) + r) * DDIM + c4]);
        *reinterpret_cast<float4*>(&ush[r * DDIM + c4]) = val;
    }

    // Preload W B-fragments once: invariant across all 8 steps (~64 VGPRs/lane).
    Frag wf[8];
#pragma unroll
    for (int kc = 0; kc < 8; ++kc) {
#pragma unroll
        for (int v = 0; v < 8; ++v) {
            const int k0 = kc * 32 + kh * 8 + 2 * v + (v >= 4 ? 8 : 0);
            wf[kc].e[2 * v]     = (__bf16)weight[(long)k0 * DDIM + ncol];
            wf[kc].e[2 * v + 1] = (__bf16)weight[(long)(k0 + 1) * DDIM + ncol];
        }
    }
    const float bias_n   = bias[ncol];
    const float invtau_n = 1.0f / tau[ncol];

    // h0 = 0 (registers in C layout + LDS bf16 copy).
    for (int i = tid; i < 16 * DDIM / 2; i += 512)
        reinterpret_cast<unsigned int*>(hsh)[i] = 0u;
    float h[8];
#pragma unroll
    for (int r = 0; r < 8; ++r) h[r] = 0.0f;
    __syncthreads();

    for (int t = 0; t < KSTEPS; ++t) {
        // acc = h @ W  (this wave's 16 output columns)
        v8f acc = {};
#pragma unroll
        for (int kc = 0; kc < 8; ++kc) {
            Frag a;
            const __bf16* src = &hsh[m * DDIM + kc * 32 + kh * 8];
            a.q[0] = *reinterpret_cast<const B128*>(src);        // K = base .. base+7
            a.q[1] = *reinterpret_cast<const B128*>(src + 16);   // K = base+16 .. base+23
            acc = __builtin_amdgcn_wmma_f32_16x16x32_bf16(
                false, a.v, false, wf[kc].v, (short)0, acc, false, false);
        }
        __syncthreads();  // all waves finished reading h before it is overwritten

#pragma unroll
        for (int r = 0; r < 8; ++r) {
            const int mr = r + 8 * kh;
            // step t consumes staged U row (mr + t): global row row0 + mr - (7 - t)
            const float u   = ush[(mr + t) * DDIM + ncol];
            const float pre = acc[r] + u + bias_n;
            const float rl  = pre > 0.0f ? pre : 0.0f;
            h[r] += invtau_n * (rl - h[r]);
            hsh[mr * DDIM + ncol] = (__bf16)h[r];
        }
        __syncthreads();  // new h visible to all waves for next step
    }

#pragma unroll
    for (int r = 0; r < 8; ++r) {
        const long grow = row0 + r + 8 * kh;
        out[grow * DDIM + ncol] = h[r];
    }
}

// ---------------------------------------------------------------------------
extern "C" void kernel_launch(void* const* d_in, const int* in_sizes, int n_in,
                              void* d_out, int out_size, void* d_ws, size_t ws_size,
                              hipStream_t stream)
{
    const float* x            = (const float*)d_in[0];  // (8, 2048, 256)
    const float* weight       = (const float*)d_in[1];  // (256, 256)
    const float* input_weight = (const float*)d_in[2];  // (256, 256)
    const float* bias         = (const float*)d_in[3];  // (1, 256)
    const float* tau          = (const float*)d_in[4];  // (1, 256)
    // d_in[5] = ksize (== 8, baked into KSTEPS)

    float* U   = (float*)d_ws;   // 16384 x 256 f32 = 16.8 MB scratch
    float* out = (float*)d_out;  // (8, 2048, 256) f32

    dim3 g1(NROW / 16, 2);
    localrnn_proj_u<<<g1, dim3(256), 0, stream>>>(x, input_weight, U);
    localrnn_recur<<<dim3(NROW / 16), dim3(512), 0, stream>>>(weight, bias, tau, U, out);
}
